// CausalSelfAttention_6768868458990
// MI455X (gfx1250) — matrix operand order repaired and checked by reading, not compiled, not run on hardware
//
#include <hip/hip_runtime.h>

// ---------------------------------------------------------------------------
// CDNA5 (gfx1250) causal self-attention:
//   qkv = x @ w_attn^T          (bf16 WMMA GEMM, f32 accum, bf16 out -> ws)
//   attn = flash_attention(qkv) (bf16 WMMA, online softmax, bf16 out -> ws)
//   out  = attn @ w_proj^T      (bf16 WMMA GEMM + TDM tensor_load_to_lds)
// B=4, S=2048, D=1024, H=16, HD=64
// ---------------------------------------------------------------------------

typedef __attribute__((ext_vector_type(16))) __bf16        v16bf;
typedef __attribute__((ext_vector_type(8)))  float         v8f;
typedef __attribute__((ext_vector_type(4)))  unsigned int  u32x4;
typedef __attribute__((ext_vector_type(4)))  int           i32x4;
typedef __attribute__((ext_vector_type(8)))  int           i32x8;

union FragB16 {
    v16bf v;
    uint4 u[2];
};

__device__ __forceinline__ unsigned short f2bf(float f) {
    union { float f; unsigned int u; } c;
    c.f = f;
    unsigned int u = c.u;
    unsigned int r = u + 0x7FFFu + ((u >> 16) & 1u);   // round-to-nearest-even
    return (unsigned short)(r >> 16);
}

__device__ __forceinline__ v8f wmma_bf16(const FragB16& a, const FragB16& b, v8f c) {
    return __builtin_amdgcn_wmma_f32_16x16x32_bf16(
        false, a.v, false, b.v, (short)0, c, false, false);
}

// Low 32 bits of a generic pointer to a __shared__ object == LDS byte offset.
__device__ __forceinline__ unsigned lds_addr_u32(const void* p) {
    return (unsigned)(unsigned long long)p;
}

#if __has_builtin(__builtin_amdgcn_tensor_load_to_lds)
#define HAVE_TDM 1
// One-shot TDM load of a [rows x 32] bf16 tile from a row-major [*, K] bf16
// tensor into LDS (contiguous 64B rows). D# per CDNA5 ISA.
__device__ __forceinline__ void tdm_load_tile_bf16(
    const void* gsrc, unsigned lds_byte_off, int K, int rows)
{
    unsigned long long ga = (unsigned long long)gsrc;
    u32x4 g0;
    g0[0] = 0x1u;                                   // count=1, user load
    g0[1] = lds_byte_off;                           // lds_addr
    g0[2] = (unsigned)ga;                           // global_addr[31:0]
    g0[3] = ((unsigned)(ga >> 32) & 0x01FFFFFFu)    // global_addr[56:32]
            | (2u << 30);                           // type=2 ("image")
    i32x8 g1;
    g1[0] = 0x00010000;                             // data_size=1 (2 bytes)
    g1[1] = (int)((K & 0xFFFF) << 16);              // tensor_dim0[15:0]
    g1[2] = (int)(((K >> 16) & 0xFFFF)              // tensor_dim0[31:16]
            | ((8192 & 0xFFFF) << 16));             // tensor_dim1[15:0]
    g1[3] = (int)(32u << 16);                       // tile_dim0 = 32 (K)
    g1[4] = rows;                                   // tile_dim1 = rows; tile_dim2=0
    g1[5] = K;                                      // tensor_dim0_stride[31:0]
    g1[6] = 0;                                      // stride hi / dim1_stride lo
    g1[7] = 0;                                      // tensor_dim1_stride hi
    i32x4 z4 = {};
#if __clang_major__ >= 23
    i32x8 z8 = {};
    __builtin_amdgcn_tensor_load_to_lds(g0, g1, z4, z4, z8, 0);
#else
    __builtin_amdgcn_tensor_load_to_lds(g0, g1, z4, z4, 0);
#endif
}
#else
#define HAVE_TDM 0
#endif

// ---------------------------------------------------------------------------
// Tiled GEMM:  C[M,N] = A[M,K] * W[N,K]^T   (W row-major [out,in], contiguous K)
// Block = 256 threads (8 waves), tile 128x128, BK=32.
// Wave (wm in 0..1, wn in 0..3) computes 64(M) x 32(N): 4 m-tiles x 2 n-tiles.
// A_BF16 path stages the A tile with the Tensor Data Mover (TDM).
// ---------------------------------------------------------------------------
template <bool A_BF16, bool OUT_BF16>
__global__ __launch_bounds__(256) void gemm_wmma_bf16(
    const void* __restrict__ Ap, const float* __restrict__ W,
    void* __restrict__ Cp, int N, int K)
{
    __shared__ unsigned short As[128 * 32];
    __shared__ unsigned short Bs[128 * 32];

    const int t    = threadIdx.x;
    const int lane = t & 31;
    const int wave = t >> 5;
    const int wm   = wave >> 2;        // 0..1  -> 64 rows
    const int wn   = wave & 3;         // 0..3  -> 32 cols
    const int bm   = blockIdx.x * 128;
    const int bn   = blockIdx.y * 128;

    const int rsel = lane & 15;
    const int off0 = (lane & 16) ? 8 : 0;   // A-frag K base (pairs)
    const int kb   = (lane & 16) ? 16 : 0;  // unused (B frag also uses off0)

    v8f acc[4][2];
#pragma unroll
    for (int i = 0; i < 4; ++i)
#pragma unroll
        for (int j = 0; j < 2; ++j) acc[i][j] = v8f{};

    const int lrow = t >> 1;           // 0..127
    const int lkh  = (t & 1) * 16;     // 0 or 16

    for (int k0 = 0; k0 < K; k0 += 32) {
        // ---- stage A tile (128 x 32) as bf16 in LDS ----
        if constexpr (A_BF16) {
#if HAVE_TDM
            if (wave == 0) {
                const unsigned short* A16 = (const unsigned short*)Ap;
                tdm_load_tile_bf16(A16 + (size_t)bm * K + k0,
                                   lds_addr_u32(&As[0]), K, 128);
                __builtin_amdgcn_s_wait_tensorcnt(0);
            }
#else
            const unsigned short* A16 = (const unsigned short*)Ap;
            const uint4* src = (const uint4*)(A16 + (size_t)(bm + lrow) * K + k0 + lkh);
            *(uint4*)&As[lrow * 32 + lkh]     = src[0];
            *(uint4*)&As[lrow * 32 + lkh + 8] = src[1];
#endif
        } else {
            const float* A32 = (const float*)Ap;
            const float* src = A32 + (size_t)(bm + lrow) * K + k0 + lkh;
            unsigned short* dst = &As[lrow * 32 + lkh];
#pragma unroll
            for (int j = 0; j < 16; ++j) dst[j] = f2bf(src[j]);
            if (k0 + 32 < K) __builtin_prefetch(src + 32, 0, 1);
        }
        // ---- stage W tile (128 n x 32 k) as bf16 in LDS ----
        {
            const float* src = W + (size_t)(bn + lrow) * K + k0 + lkh;
            unsigned short* dst = &Bs[lrow * 32 + lkh];
#pragma unroll
            for (int j = 0; j < 16; ++j) dst[j] = f2bf(src[j]);
            if (k0 + 32 < K) __builtin_prefetch(src + 32, 0, 1);
        }
        __syncthreads();

        // ---- build fragments (ISA 16-bit A 16x32 / B 32x16 layouts) ----
        FragB16 af[4], bf[2];
#pragma unroll
        for (int mt = 0; mt < 4; ++mt) {
            const unsigned short* p = &As[(wm * 64 + mt * 16 + rsel) * 32];
            af[mt].u[0] = *(const uint4*)(p + off0);
            af[mt].u[1] = *(const uint4*)(p + off0 + 16);
        }
#pragma unroll
        for (int nt = 0; nt < 2; ++nt) {
            const unsigned short* p = &Bs[(wn * 32 + nt * 16 + rsel) * 32 + off0];
            bf[nt].u[0] = *(const uint4*)(p);
            bf[nt].u[1] = *(const uint4*)(p + 16);
        }
#pragma unroll
        for (int mt = 0; mt < 4; ++mt)
#pragma unroll
            for (int nt = 0; nt < 2; ++nt)
                acc[mt][nt] = wmma_bf16(af[mt], bf[nt], acc[mt][nt]);

        __syncthreads();
    }

    // ---- epilogue: C layout VGPR r -> row r (lanes 0-15) / 8+r (16-31) ----
    const int rowhi = (lane & 16) ? 8 : 0;
#pragma unroll
    for (int mt = 0; mt < 4; ++mt)
#pragma unroll
        for (int nt = 0; nt < 2; ++nt)
#pragma unroll
            for (int r = 0; r < 8; ++r) {
                const int grow = bm + wm * 64 + mt * 16 + r + rowhi;
                const int gcol = bn + wn * 32 + nt * 16 + rsel;
                const float val = acc[mt][nt][r];
                if constexpr (OUT_BF16)
                    ((unsigned short*)Cp)[(size_t)grow * N + gcol] = f2bf(val);
                else
                    ((float*)Cp)[(size_t)grow * N + gcol] = val;
            }
}

// ---------------------------------------------------------------------------
// Flash attention: one wave per 16-row q-tile, 32-key chunks, online softmax.
// qkv: bf16 [B*S, 3072] rows = [q(1024) | k(1024) | v(1024)]; out bf16 [B*S,1024]
// ---------------------------------------------------------------------------
__global__ __launch_bounds__(128) void flash_attn_wmma(
    const unsigned short* __restrict__ qkv, unsigned short* __restrict__ outw)
{
    constexpr int S = 2048, HD = 64, RS = 3072;
    __shared__ unsigned short Vtr[4][64 * 32];  // per wave: [hd][key]
    __shared__ unsigned short Pb[4][16 * 32];   // per wave: [qrow][key]

    const int lane  = threadIdx.x & 31;
    const int wave  = threadIdx.x >> 5;
    const int qtile = blockIdx.x * 4 + wave;
    const int qbase = qtile * 16;
    const int bh    = blockIdx.y;
    const int b     = bh >> 4;
    const int h     = bh & 15;

    const size_t rowbase = (size_t)b * S * RS;
    const int qcol = h * HD;
    const int kcol = 1024 + h * HD;
    const int vcol = 2048 + h * HD;

    unsigned short* vtr = Vtr[wave];
    unsigned short* pb  = Pb[wave];

    const int rsel = lane & 15;
    const int hi   = (lane & 16) ? 1 : 0;
    const int off0 = hi ? 8 : 0;
    const int kb   = hi ? 16 : 0;

    // Q fragments (A layout), hd split into two K=32 halves
    FragB16 qf[2];
    {
        const unsigned short* qp = qkv + rowbase + (size_t)(qbase + rsel) * RS + qcol;
#pragma unroll
        for (int f = 0; f < 2; ++f) {
            qf[f].u[0] = *(const uint4*)(qp + f * 32 + off0);
            qf[f].u[1] = *(const uint4*)(qp + f * 32 + off0 + 16);
        }
    }

    v8f o[4];
#pragma unroll
    for (int f = 0; f < 4; ++f) o[f] = v8f{};
    float m[8], l[8];
#pragma unroll
    for (int r = 0; r < 8; ++r) { m[r] = -1e30f; l[r] = 0.0f; }

    const int kend = qbase + 16;           // causal bound (exclusive)
    for (int kc = 0; kc < kend; kc += 32) {
        // ---- scores: S = Q K^T (two 16x16 key tiles, two hd halves each) ----
        v8f st[2];
#pragma unroll
        for (int tk = 0; tk < 2; ++tk) {
            const unsigned short* kp =
                qkv + rowbase + (size_t)(kc + tk * 16 + rsel) * RS + kcol;
            FragB16 kf0, kf1;                       // B layout: K = hd
            kf0.u[0] = *(const uint4*)(kp + off0);
            kf0.u[1] = *(const uint4*)(kp + off0 + 16);
            kf1.u[0] = *(const uint4*)(kp + 32 + off0);
            kf1.u[1] = *(const uint4*)(kp + 32 + off0 + 16);
            v8f z = v8f{};
            st[tk] = wmma_bf16(qf[0], kf0, z);
            st[tk] = wmma_bf16(qf[1], kf1, st[tk]);
        }

        // ---- scale (+ causal mask only on the diagonal-straddling chunk) ----
        if (kc + 31 > qbase) {
#pragma unroll
            for (int tk = 0; tk < 2; ++tk)
#pragma unroll
                for (int r = 0; r < 8; ++r) {
                    const float s  = st[tk][r] * 0.125f;   // 1/sqrt(64)
                    const int qrow = qbase + r + hi * 8;
                    const int kcg  = kc + tk * 16 + rsel;
                    st[tk][r] = (kcg <= qrow) ? s : -1e30f;
                }
        } else {
#pragma unroll
            for (int tk = 0; tk < 2; ++tk)
#pragma unroll
                for (int r = 0; r < 8; ++r)
                    st[tk][r] *= 0.125f;
        }

        // ---- online softmax: row max over 32 keys (xor-reduce 16-lane group) ----
        float mx[8];
#pragma unroll
        for (int r = 0; r < 8; ++r) {
            mx[r] = fmaxf(st[0][r], st[1][r]);
#pragma unroll
            for (int d = 1; d <= 8; d <<= 1)
                mx[r] = fmaxf(mx[r], __shfl_xor(mx[r], d, 32));
        }
        float alpha[8];
#pragma unroll
        for (int r = 0; r < 8; ++r) {
            const float mn = fmaxf(m[r], mx[r]);
            alpha[r] = __expf(m[r] - mn);
            m[r] = mn;
        }
#pragma unroll
        for (int tk = 0; tk < 2; ++tk)
#pragma unroll
            for (int r = 0; r < 8; ++r)
                st[tk][r] = __expf(st[tk][r] - m[r]);
#pragma unroll
        for (int r = 0; r < 8; ++r) {
            float sr = st[0][r] + st[1][r];
#pragma unroll
            for (int d = 1; d <= 8; d <<= 1)
                sr += __shfl_xor(sr, d, 32);
            l[r] = l[r] * alpha[r] + sr;
        }
#pragma unroll
        for (int f = 0; f < 4; ++f)
#pragma unroll
            for (int r = 0; r < 8; ++r) o[f][r] *= alpha[r];

        // ---- P (C layout) -> bf16 LDS [qrow][key] for A-layout reload ----
#pragma unroll
        for (int tk = 0; tk < 2; ++tk)
#pragma unroll
            for (int r = 0; r < 8; ++r)
                pb[(r + hi * 8) * 32 + tk * 16 + rsel] = f2bf(st[tk][r]);

        // ---- V chunk [32 keys][64 hd] -> LDS transposed [hd][key] ----
        {
            const uint4* vp =
                (const uint4*)(qkv + rowbase + (size_t)(kc + lane) * RS + vcol);
#pragma unroll
            for (int c = 0; c < 8; ++c) {
                uint4 d4 = vp[c];
                const unsigned short* e = (const unsigned short*)&d4;
#pragma unroll
                for (int j = 0; j < 8; ++j)
                    vtr[(c * 8 + j) * 32 + lane] = e[j];
            }
        }
        asm volatile("s_wait_dscnt 0" ::: "memory");   // wave-local store->load

        // ---- O += P V : K = 32 keys, 4 hd tiles of N=16 ----
        FragB16 pf;
        {
            const unsigned short* pp = &pb[rsel * 32];
            pf.u[0] = *(const uint4*)(pp + off0);
            pf.u[1] = *(const uint4*)(pp + off0 + 16);
        }
#pragma unroll
        for (int f = 0; f < 4; ++f) {
            FragB16 vf;
            const unsigned short* vv = &vtr[(f * 16 + rsel) * 32 + off0];
            vf.u[0] = *(const uint4*)(vv);
            vf.u[1] = *(const uint4*)(vv + 16);
            o[f] = wmma_bf16(pf, vf, o[f]);
        }
        asm volatile("s_wait_dscnt 0" ::: "memory");   // before next-iter stores
    }

    // ---- finalize: O / l, write bf16 [b*S+s, h*64+hd] ----
#pragma unroll
    for (int f = 0; f < 4; ++f)
#pragma unroll
        for (int r = 0; r < 8; ++r) {
            const int qrow = qbase + r + hi * 8;
            const int col  = h * HD + f * 16 + rsel;
            const float val = o[f][r] * (1.0f / l[r]);
            outw[(size_t)(b * S + qrow) * 1024 + col] = f2bf(val);
        }
}

// ---------------------------------------------------------------------------
extern "C" void kernel_launch(void* const* d_in, const int* in_sizes, int n_in,
                              void* d_out, int out_size, void* d_ws, size_t ws_size,
                              hipStream_t stream) {
    (void)in_sizes; (void)n_in; (void)out_size;
    constexpr int B = 4, S = 2048, D = 1024, H = 16;
    constexpr size_t M = (size_t)B * S;          // 8192 rows

    const float* x      = (const float*)d_in[0];
    const float* w_attn = (const float*)d_in[1];
    const float* w_proj = (const float*)d_in[2];
    float*       out    = (float*)d_out;

    unsigned short* qkv  = (unsigned short*)d_ws;            // [M, 3D] bf16
    unsigned short* attn = qkv + M * (size_t)(3 * D);        // [M, D]  bf16
    const size_t need = (M * (size_t)(3 * D) + M * (size_t)D) * sizeof(unsigned short);
    if (ws_size < need) return;

    // 1) qkv = x @ w_attn^T   (M=8192, N=3072, K=1024)
    gemm_wmma_bf16<false, true><<<dim3(M / 128, (3 * D) / 128), 256, 0, stream>>>(
        x, w_attn, qkv, 3 * D, D);

    // 2) flash attention over qkv -> attn
    flash_attn_wmma<<<dim3(S / 64, B * H), 128, 0, stream>>>(qkv, attn);

    // 3) out = attn @ w_proj^T  (M=8192, N=1024, K=1024) with TDM A-staging
    gemm_wmma_bf16<true, false><<<dim3(M / 128, D / 128), 256, 0, stream>>>(
        attn, w_proj, out, D, D);
}
